// GRU_27599459844841
// MI455X (gfx1250) — matrix operand order repaired and checked by reading, not compiled, not run on hardware
//
#include <hip/hip_runtime.h>
#include <hip/hip_bf16.h>

// ---------------------------------------------------------------------------
// GRU on MI455X (gfx1250):
//   K1: gi = x(1024x32000) @ w_ih^T(32000x768) + b_ih  -- bf16 WMMA, fp32 acc,
//       fp32->bf16 conversion fused into LDS staging, double-buffered LDS,
//       one barrier per K-step, next-tile loads issued before compute.
//   K2: persistent single-workgroup recurrence, w_hh resident in VGPRs as
//       pre-swizzled WMMA A-fragments; 16 v_wmma per wave per step.
//   K3: out[t] = hs[t,:] . w_lin + b_lin
// ---------------------------------------------------------------------------

typedef __attribute__((ext_vector_type(16))) __bf16 v16bf;
typedef __attribute__((ext_vector_type(8)))  float  v8f;
typedef __attribute__((ext_vector_type(4)))  __bf16 bf16x4;

union Frag16 { v16bf v; uint4 q[2]; };

#define GM 1024
#define GN 768
#define GK 32000
#define BM 64
#define BN 64
#define BK 32
#define NK (GK / BK)   // 1000 K-steps
#define LDSS 40        // LDS row stride in bf16 elems (80B): 16B-aligned, bank-spread

// ============================ Kernel 1: big GEMM ===========================
__global__ __launch_bounds__(128) void gemm_gi(const float* __restrict__ x,
                                               const float* __restrict__ w_ih,
                                               const float* __restrict__ b_ih,
                                               float* __restrict__ gi) {
  __shared__ __align__(16) __bf16 As[2][BM * LDSS];
  __shared__ __align__(16) __bf16 Bs[2][BN * LDSS];

  const int tid  = threadIdx.x;
  const int lane = tid & 31;
  const int wave = tid >> 5;             // 0..3
  const int wm   = (wave >> 1) * 32;     // wave M offset in tile
  const int wn   = (wave & 1) * 32;      // wave N offset in tile
  const int m0   = blockIdx.y * BM;
  const int n0   = blockIdx.x * BN;
  const int lh   = lane >> 4;            // lane half (0/1)
  const int lr   = lane & 15;

  v8f acc[2][2] = {};

  // per-thread staging coordinates: 512 float4 quads over 128 threads
  int srow[4], skq[4];
#pragma unroll
  for (int i = 0; i < 4; ++i) {
    int q   = tid + i * 128;
    srow[i] = q >> 3;
    skq[i]  = (q & 7) * 4;
  }

  // ---- prologue: stage tile 0 into buffer 0 ----
#pragma unroll
  for (int i = 0; i < 4; ++i) {
    float4 a = *(const float4*)(x    + (size_t)(m0 + srow[i]) * GK + skq[i]);
    float4 b = *(const float4*)(w_ih + (size_t)(n0 + srow[i]) * GK + skq[i]);
    bf16x4 av = {(__bf16)a.x, (__bf16)a.y, (__bf16)a.z, (__bf16)a.w};
    bf16x4 bv = {(__bf16)b.x, (__bf16)b.y, (__bf16)b.z, (__bf16)b.w};
    *(bf16x4*)&As[0][srow[i] * LDSS + skq[i]] = av;
    *(bf16x4*)&Bs[0][srow[i] * LDSS + skq[i]] = bv;
  }
  __syncthreads();

  for (int it = 0; it < NK; ++it) {
    const int cur  = it & 1;
    const int nxt  = cur ^ 1;
    const bool more = (it + 1) < NK;

    // ---- issue global loads of tile it+1 first (latency under WMMA) ----
    float4 an[4], bn[4];
    if (more) {
      const int k1 = (it + 1) * BK;
#pragma unroll
      for (int i = 0; i < 4; ++i) {
        const float* pa = x    + (size_t)(m0 + srow[i]) * GK + k1 + skq[i];
        const float* pb = w_ih + (size_t)(n0 + srow[i]) * GK + k1 + skq[i];
        an[i] = *(const float4*)pa;
        bn[i] = *(const float4*)pb;
        if ((it + 2) < NK) {             // prefetch tile it+2 into GL2
          __builtin_prefetch(pa + BK, 0, 0);
          __builtin_prefetch(pb + BK, 0, 0);
        }
      }
    }

    // ---- fragments from LDS buffer `cur` (per ISA 7.12.2 layouts) ----
    Frag16 a[2], b[2];
#pragma unroll
    for (int mi = 0; mi < 2; ++mi) {
      // A 16x32: lane<16 holds K{0..7,16..23}, lane>=16 holds K{8..15,24..31}
      const __bf16* p = &As[cur][(wm + mi * 16 + lr) * LDSS + lh * 8];
      a[mi].q[0] = *(const uint4*)p;
      a[mi].q[1] = *(const uint4*)(p + 16);
    }
#pragma unroll
    for (int ni = 0; ni < 2; ++ni) {
      // B 32x16: lane<16 holds N=lane K{0..7,16..23}, lane>=16 holds K{8..15,24..31}
      const __bf16* p = &Bs[cur][(wn + ni * 16 + lr) * LDSS + lh * 8];
      b[ni].q[0] = *(const uint4*)p;
      b[ni].q[1] = *(const uint4*)(p + 16);
    }
#pragma unroll
    for (int mi = 0; mi < 2; ++mi)
#pragma unroll
      for (int ni = 0; ni < 2; ++ni)
        acc[mi][ni] = __builtin_amdgcn_wmma_f32_16x16x32_bf16(
            false, a[mi].v, false, b[ni].v, (short)0, acc[mi][ni], false, false);

    // ---- convert + store tile it+1 into buffer `nxt` (disjoint from cur) ----
    if (more) {
#pragma unroll
      for (int i = 0; i < 4; ++i) {
        bf16x4 av = {(__bf16)an[i].x, (__bf16)an[i].y, (__bf16)an[i].z, (__bf16)an[i].w};
        bf16x4 bv = {(__bf16)bn[i].x, (__bf16)bn[i].y, (__bf16)bn[i].z, (__bf16)bn[i].w};
        *(bf16x4*)&As[nxt][srow[i] * LDSS + skq[i]] = av;
        *(bf16x4*)&Bs[nxt][srow[i] * LDSS + skq[i]] = bv;
      }
    }
    __syncthreads();   // single barrier per K-step
  }

  // ---- epilogue: D layout -> gi, add input bias ----
#pragma unroll
  for (int mi = 0; mi < 2; ++mi)
#pragma unroll
    for (int ni = 0; ni < 2; ++ni) {
      int col  = n0 + wn + ni * 16 + lr;
      int rowb = m0 + wm + mi * 16 + lh * 8;
      float bias = b_ih[col];
#pragma unroll
      for (int r = 0; r < 8; ++r)
        gi[(size_t)(rowb + r) * GN + col] = acc[mi][ni][r] + bias;
    }
}

// ===================== Kernel 2: persistent recurrence =====================
// 768 threads = 24 waves; wave w owns gh rows [32w, 32w+32) as two 16-row
// WMMA M-tiles. w_hh lives in VGPRs as bf16 A-fragments for all 1024 steps.
__global__ __launch_bounds__(768) void gru_rec(const float* __restrict__ gi,
                                               const float* __restrict__ w_hh,
                                               const float* __restrict__ b_hh,
                                               float* __restrict__ hs) {
  __shared__ float h_f[256];
  __shared__ __align__(16) __bf16 h_bf[256];
  __shared__ float gh[768];

  const int tid  = threadIdx.x;
  const int lane = tid & 31;
  const int wave = tid >> 5;     // 0..23
  const int lh   = lane >> 4;
  const int lr   = lane & 15;

  // ---- preload w_hh into A-fragment registers (bf16) ----
  Frag16 wf[2][8];
#pragma unroll
  for (int mt = 0; mt < 2; ++mt) {
    const float* wr = w_hh + (size_t)(wave * 32 + mt * 16 + lr) * 256;
#pragma unroll
    for (int kt = 0; kt < 8; ++kt) {
      int kb = kt * 32 + lh * 8;
      float4 f0 = *(const float4*)(wr + kb);
      float4 f1 = *(const float4*)(wr + kb + 4);
      float4 f2 = *(const float4*)(wr + kb + 16);
      float4 f3 = *(const float4*)(wr + kb + 20);
      float fl[16] = {f0.x, f0.y, f0.z, f0.w, f1.x, f1.y, f1.z, f1.w,
                      f2.x, f2.y, f2.z, f2.w, f3.x, f3.y, f3.z, f3.w};
      v16bf v;
#pragma unroll
      for (int i = 0; i < 16; ++i) v[i] = (__bf16)fl[i];
      wf[mt][kt].v = v;
    }
  }

  // hidden-side bias held in registers by gate threads
  float br = 0.f, bz = 0.f, bn = 0.f;
  if (tid < 256) {
    br = b_hh[tid]; bz = b_hh[tid + 256]; bn = b_hh[tid + 512];
    h_f[tid] = 0.f;
    h_bf[tid] = (__bf16)0.f;
  }
  __syncthreads();

  for (int t = 0; t < 1024; ++t) {
    // issue gi[t] loads early; consumed after the barrier (latency overlap)
    float gir = 0.f, giz = 0.f, gin = 0.f;
    if (tid < 256) {
      const float* g = gi + (size_t)t * GN;
      gir = g[tid]; giz = g[tid + 256]; gin = g[tid + 512];
    }

    // ---- gh = w_hh @ h via WMMA (h broadcast into N=0 column of B) ----
    v8f c0 = {}, c1 = {};
#pragma unroll
    for (int kt = 0; kt < 8; ++kt) {
      const uint4* hp = (const uint4*)&h_bf[kt * 32 + lh * 8];
      uint4 q0 = hp[0], q1 = hp[2];
      uint4 z  = {0u, 0u, 0u, 0u};
      Frag16 b;
      b.q[0] = (lr == 0) ? q0 : z;   // only N=0 column is live
      b.q[1] = (lr == 0) ? q1 : z;
      c0 = __builtin_amdgcn_wmma_f32_16x16x32_bf16(false, wf[0][kt].v, false,
                                                   b.v, (short)0, c0, false, false);
      c1 = __builtin_amdgcn_wmma_f32_16x16x32_bf16(false, wf[1][kt].v, false,
                                                   b.v, (short)0, c1, false, false);
    }
    // D layout: lane0 VGPR r -> (M=r,N=0); lane16 -> (M=8+r,N=0)
    if (lr == 0) {
      int b0 = wave * 32 + lh * 8;
      int b1 = wave * 32 + 16 + lh * 8;
#pragma unroll
      for (int r = 0; r < 8; ++r) { gh[b0 + r] = c0[r]; gh[b1 + r] = c1[r]; }
    }
    __syncthreads();

    // ---- gates + state update on threads 0..255 ----
    if (tid < 256) {
      float ghr = gh[tid] + br;
      float ghz = gh[tid + 256] + bz;
      float ghn = gh[tid + 512] + bn;
      float r = 1.f / (1.f + __expf(-(gir + ghr)));
      float z = 1.f / (1.f + __expf(-(giz + ghz)));
      float n = tanhf(gin + r * ghn);
      float hnew = (1.f - z) * n + z * h_f[tid];
      h_f[tid]  = hnew;
      h_bf[tid] = (__bf16)hnew;
      hs[(size_t)t * 256 + tid] = hnew;
    }
    __syncthreads();
  }
}

// ======================= Kernel 3: output projection =======================
__global__ __launch_bounds__(256) void final_lin(const float* __restrict__ hs,
                                                 const float* __restrict__ w_lin,
                                                 const float* __restrict__ b_lin,
                                                 float* __restrict__ out) {
  __shared__ float red[8];
  const int t = blockIdx.x;
  float p = hs[(size_t)t * 256 + threadIdx.x] * w_lin[threadIdx.x];
#pragma unroll
  for (int off = 16; off > 0; off >>= 1) p += __shfl_down(p, off, 32);
  if ((threadIdx.x & 31) == 0) red[threadIdx.x >> 5] = p;
  __syncthreads();
  if (threadIdx.x == 0) {
    float s = 0.f;
#pragma unroll
    for (int i = 0; i < 8; ++i) s += red[i];
    out[t] = s + b_lin[0];
  }
}

// =============================== launcher ==================================
extern "C" void kernel_launch(void* const* d_in, const int* in_sizes, int n_in,
                              void* d_out, int out_size, void* d_ws, size_t ws_size,
                              hipStream_t stream) {
  const float* x     = (const float*)d_in[0];
  const float* w_ih  = (const float*)d_in[1];
  const float* w_hh  = (const float*)d_in[2];
  const float* b_ih  = (const float*)d_in[3];
  const float* b_hh  = (const float*)d_in[4];
  const float* w_lin = (const float*)d_in[5];
  const float* b_lin = (const float*)d_in[6];
  float* out = (float*)d_out;

  float* gi = (float*)d_ws;                    // 1024*768 f32 = 3 MB
  float* hs = gi + (size_t)GM * GN;            // 1024*256 f32 = 1 MB

  gemm_gi<<<dim3(GN / BN, GM / BM), 128, 0, stream>>>(x, w_ih, b_ih, gi);
  gru_rec<<<1, 768, 0, stream>>>(gi, w_hh, b_hh, hs);
  final_lin<<<GM, 256, 0, stream>>>(hs, w_lin, b_lin, out);
}
